// DAWN_74526272520648
// MI455X (gfx1250) — matrix operand order repaired and checked by reading, not compiled, not run on hardware
//
#include <hip/hip_runtime.h>

typedef __attribute__((ext_vector_type(16))) _Float16 v16h;
typedef __attribute__((ext_vector_type(8)))  _Float16 v8h;
typedef __attribute__((ext_vector_type(8)))  float    v8f;

#define S_LEN   1024
#define D_MODEL 1024
#define P_DIM   32
#define D_HEAD  64

// 32 bytes: one WMMA 16-bit fragment per lane (A or B), as vectors or raw dwords
union B16 {
  uint4 u[2];
  v8h   h8[2];
  v16h  h;
};

// ---------------------------------------------------------------- helpers

__device__ inline float wave_sum32(float v) {
#pragma unroll
  for (int m = 16; m >= 1; m >>= 1) v += __shfl_xor(v, m, 32);
  return v;
}
// reductions confined to 16-lane halves (match WMMA C-fragment halves)
__device__ inline float half_max16(float v) {
#pragma unroll
  for (int m = 1; m < 16; m <<= 1) v = fmaxf(v, __shfl_xor(v, m, 32));
  return v;
}
__device__ inline float half_sum16(float v) {
#pragma unroll
  for (int m = 1; m < 16; m <<= 1) v += __shfl_xor(v, m, 32);
  return v;
}

__device__ inline v8f wmma_f16(v16h a, v16h b, v8f c) {
  return __builtin_amdgcn_wmma_f32_16x16x32_f16(
      /*neg_a=*/false, a, /*neg_b=*/false, b,
      /*c_mod=*/(short)0, c, /*reuse_a=*/false, /*reuse_b=*/false);
}

// pack 8 f32 (two float4) into v16h elements [o, o+8)
__device__ inline void pack8(v16h& d, int o, float4 u, float4 v) {
  d[o + 0] = (_Float16)u.x; d[o + 1] = (_Float16)u.y;
  d[o + 2] = (_Float16)u.z; d[o + 3] = (_Float16)u.w;
  d[o + 4] = (_Float16)v.x; d[o + 5] = (_Float16)v.y;
  d[o + 6] = (_Float16)v.z; d[o + 7] = (_Float16)v.w;
}

// ---------------------------------------------------------------- loss init

extern "C" __global__ void dawn_zero(float* p) {
  if (threadIdx.x == 0 && blockIdx.x == 0) p[0] = 0.f;
}

// ---------------------------------------------------------------- top-k candidate select
// one block per token; distances in dynamic LDS; iterative argmin extraction

extern "C" __global__ __launch_bounds__(128)
void dawn_topk(const float* __restrict__ pos, const float* __restrict__ npos,
               int N, int ncand, int* __restrict__ idx_out, float* __restrict__ dist_out) {
  extern __shared__ float dist[];
  __shared__ float redv[128];
  __shared__ int   redi[128];
  const int token = blockIdx.x, tid = threadIdx.x;
  float p[P_DIM];
#pragma unroll
  for (int j = 0; j < P_DIM; ++j) p[j] = pos[token * P_DIM + j];
  for (int n = tid; n < N; n += 128) {
    const float* np = npos + (size_t)n * P_DIM;
    float d = 0.f;
#pragma unroll
    for (int j = 0; j < P_DIM; ++j) { const float t = p[j] - np[j]; d += t * t; }
    dist[n] = d;
  }
  __syncthreads();
  for (int k = 0; k < ncand; ++k) {
    float bv = 3.0e38f; int bi = 0;
    for (int n = tid; n < N; n += 128)
      if (dist[n] < bv) { bv = dist[n]; bi = n; }
    redv[tid] = bv; redi[tid] = bi;
    __syncthreads();
    for (int s = 64; s > 0; s >>= 1) {
      if (tid < s && redv[tid + s] < redv[tid]) { redv[tid] = redv[tid + s]; redi[tid] = redi[tid + s]; }
      __syncthreads();
    }
    if (tid == 0) {
      idx_out[token * ncand + k]  = redi[0];
      dist_out[token * ncand + k] = redv[0];
      dist[redi[0]] = 3.4e38f;
    }
    __syncthreads();
  }
}

// ---------------------------------------------------------------- router (gate + weighted sum)
// one block (256 threads) per token

template <int NC, int KEEP, bool TWO, bool RESID>
__global__ __launch_bounds__(256)
void dawn_router(const float* __restrict__ xin, const int* __restrict__ idx,
                 const float* __restrict__ dist,
                 const float* __restrict__ tau1, const float* __restrict__ tau2,
                 const float* __restrict__ neurons,
                 _Float16* __restrict__ out1, _Float16* __restrict__ out2,
                 float* __restrict__ outf, const float* __restrict__ resid,
                 float* __restrict__ loss) {
  __shared__ float xs[D_MODEL];
  __shared__ float act[NC], e1[NC], e2[NC], k1[NC], k2[NC], w1[NC], w2[NC], lv[NC];
  __shared__ float scal[4];
  const int token = blockIdx.x, tid = threadIdx.x;
  const int wave = tid >> 5, lane = tid & 31;
  for (int d = tid; d < D_MODEL; d += 256) xs[d] = xin[(size_t)token * D_MODEL + d];
  __syncthreads();
  // activations: act[n] = x . cand[n]  (neuron tables are L2-resident; prefetch one ahead)
  for (int n = wave; n < NC; n += 8) {
    if (n + 8 < NC)
      __builtin_prefetch(neurons + (size_t)idx[token * NC + n + 8] * D_MODEL + (lane << 5), 0, 0);
    const float* nr = neurons + (size_t)idx[token * NC + n] * D_MODEL;
    float s = 0.f;
    for (int i = lane; i < D_MODEL; i += 32) s += xs[i] * nr[i];
    s = wave_sum32(s);
    if (lane == 0) act[n] = s;
  }
  __syncthreads();
  // threshold gate: eg = exp(where(raw>0, raw, 1e-8*exp(raw))) - 1
  if (tid < NC) {
    const float a = act[tid];
    {
      const float raw = a - tau1[token];
      const float g = raw > 0.f ? raw : 1e-8f * __expf(raw);
      e1[tid] = __expf(g) - 1.f;
    }
    if (TWO) {
      const float raw = a - tau2[token];
      const float g = raw > 0.f ? raw : 1e-8f * __expf(raw);
      e2[tid] = __expf(g) - 1.f;
    }
  }
  __syncthreads();
  // keep iff fewer than KEEP strictly-greater values (== eg >= kth largest)
  if (tid < NC) {
    int c1 = 0, c2 = 0;
    const float v1 = e1[tid], v2 = TWO ? e2[tid] : 0.f;
    for (int j = 0; j < NC; ++j) {
      c1 += e1[j] > v1;
      if (TWO) c2 += e2[j] > v2;
    }
    k1[tid] = (c1 < KEEP) ? v1 : 0.f;
    if (TWO) k2[tid] = (c2 < KEEP) ? v2 : 0.f;
  }
  __syncthreads();
  if (tid == 0) {
    float s1 = 0.f, m1 = 0.f, s2 = 0.f, m2 = 0.f;
    for (int j = 0; j < NC; ++j) {
      s1 += k1[j]; m1 = fmaxf(m1, k1[j]);
      if (TWO) { s2 += k2[j]; m2 = fmaxf(m2, k2[j]); }
    }
    scal[0] = s1; scal[1] = m1; scal[2] = s2; scal[3] = m2;
  }
  __syncthreads();
  if (tid < NC) {
    const float g1 = k1[tid] / (scal[0] + 1e-8f) * tanhf(scal[1]);
    w1[tid] = act[tid] * g1;
    lv[tid] = g1 * dist[token * NC + tid];
    if (TWO) {
      const float g2 = k2[tid] / (scal[2] + 1e-8f) * tanhf(scal[3]);
      w2[tid] = act[tid] * g2;
    }
  }
  __syncthreads();
  if (tid == 0) {
    float s = 0.f;
    for (int j = 0; j < NC; ++j) s += lv[j];
    atomicAdd(loss, s * (1.f / (float)(S_LEN * NC)));
  }
  // weighted candidate sum: each thread owns 4 output columns
  const int d0 = tid * 4;
  float a1[4] = {0.f, 0.f, 0.f, 0.f}, a2[4] = {0.f, 0.f, 0.f, 0.f};
  for (int n = 0; n < NC; ++n) {
    const float4 v = *(const float4*)(neurons + (size_t)idx[token * NC + n] * D_MODEL + d0);
    const float u1 = w1[n];
    a1[0] += u1 * v.x; a1[1] += u1 * v.y; a1[2] += u1 * v.z; a1[3] += u1 * v.w;
    if (TWO) {
      const float u2 = w2[n];
      a2[0] += u2 * v.x; a2[1] += u2 * v.y; a2[2] += u2 * v.z; a2[3] += u2 * v.w;
    }
  }
  if (RESID) {
#pragma unroll
    for (int c = 0; c < 4; ++c)
      outf[(size_t)token * D_MODEL + d0 + c] = resid[(size_t)token * D_MODEL + d0 + c] + a1[c];
  } else {
#pragma unroll
    for (int c = 0; c < 4; ++c) out1[(size_t)token * D_MODEL + d0 + c] = (_Float16)a1[c];
    if (TWO) {
#pragma unroll
      for (int c = 0; c < 4; ++c) out2[(size_t)token * D_MODEL + d0 + c] = (_Float16)a2[c];
    }
  }
}

// ---------------------------------------------------------------- causal flash attention (WMMA)
// grid (S/64, H); 4 waves/block, each wave owns 16 query rows; 32-key LDS tiles.
// K/V tiles staged with async global->LDS b128; K^T fragments via plain b128 LDS row loads.

extern "C" __global__ __launch_bounds__(128)
void dawn_flash_attn(const _Float16* __restrict__ Qh, const _Float16* __restrict__ Kh,
                     const _Float16* __restrict__ Vh, float* __restrict__ O) {
  __shared__ alignas(16) _Float16 Kt[32 * D_HEAD];
  __shared__ alignas(16) _Float16 Vt[32 * D_HEAD];
  __shared__ alignas(16) _Float16 Pst[4 * 16 * 32];
  const int base = blockIdx.x * 64;
  const int head = blockIdx.y;
  const int tid = threadIdx.x, wave = tid >> 5, lane = tid & 31;
  const int qb = base + wave * 16;
  const int row16 = lane & 15, half = lane >> 4;
  const unsigned ldsK = (unsigned)(size_t)&Kt[0];
  const unsigned ldsV = (unsigned)(size_t)&Vt[0];

  // Q A-fragments: per lane two contiguous 16B runs (K 0..7/16..23 or 8..15/24..31)
  B16 q0, q1;
  {
    const _Float16* qp = Qh + (size_t)(qb + row16) * D_MODEL + head * D_HEAD + (half ? 8 : 0);
    q0.u[0] = *(const uint4*)(qp);
    q0.u[1] = *(const uint4*)(qp + 16);
    q1.u[0] = *(const uint4*)(qp + 32);
    q1.u[1] = *(const uint4*)(qp + 48);
  }

  float m[8], l[8];
  v8f acc[4];
#pragma unroll
  for (int r = 0; r < 8; ++r) { m[r] = -1e30f; l[r] = 0.f; }
#pragma unroll
  for (int t = 0; t < 4; ++t)
#pragma unroll
    for (int r = 0; r < 8; ++r) acc[t][r] = 0.f;

  const int ntiles = (base + 64) / 32;
  for (int jt = 0; jt < ntiles; ++jt) {
    const int j = jt * 32;
    __syncthreads();
    // async 16B copies global -> LDS (ASYNCcnt-tracked, no VGPR round trip)
    for (int i = tid; i < 256; i += 128) {
      const int kr = i >> 3, q = i & 7;
      const unsigned lofs = (unsigned)(kr * 128 + q * 16);
      const unsigned lk = ldsK + lofs, lv = ldsV + lofs;
      const unsigned long long gk =
          (unsigned long long)(size_t)(Kh + (size_t)(j + kr) * D_MODEL + head * D_HEAD + q * 8);
      const unsigned long long gv =
          (unsigned long long)(size_t)(Vh + (size_t)(j + kr) * D_MODEL + head * D_HEAD + q * 8);
      asm volatile("global_load_async_to_lds_b128 %0, %1, off" :: "v"(lk), "v"(gk) : "memory");
      asm volatile("global_load_async_to_lds_b128 %0, %1, off" :: "v"(lv), "v"(gv) : "memory");
    }
    asm volatile("s_wait_asynccnt 0" ::: "memory");
    __syncthreads();

    // scores: S[16 x 32 keys] = Q(16x64) . K^T; K^T fragments via plain LDS row loads
    B16 bf[2][2];  // [key-half nh][d-chunk kc]
#pragma unroll
    for (int nh = 0; nh < 2; ++nh)
#pragma unroll
      for (int kc = 0; kc < 2; ++kc)
#pragma unroll
        for (int dc = 0; dc < 2; ++dc) {
          const _Float16* kp = Kt + (nh * 16 + row16) * D_HEAD + kc * 32 + dc * 16 + half * 8;
          bf[nh][kc].h8[dc] = *(const v8h*)kp;
        }
    asm volatile("s_wait_dscnt 0" ::: "memory");

    v8f s0, s1;
#pragma unroll
    for (int r = 0; r < 8; ++r) { s0[r] = 0.f; s1[r] = 0.f; }
    s0 = wmma_f16(q0.h, bf[0][0].h, s0);
    s0 = wmma_f16(q1.h, bf[0][1].h, s0);
    s1 = wmma_f16(q0.h, bf[1][0].h, s1);
    s1 = wmma_f16(q1.h, bf[1][1].h, s1);

    // scale + causal mask + online softmax (C layout: row=r+8*half, col=lane&15)
    _Float16* P = Pst + wave * (16 * 32);
#pragma unroll
    for (int r = 0; r < 8; ++r) {
      const int qrow = qb + r + 8 * half;
      float v0 = s0[r] * 0.125f;               // 1/sqrt(64)
      float v1 = s1[r] * 0.125f;
      if (j + row16 > qrow)      v0 = -1e30f;
      if (j + 16 + row16 > qrow) v1 = -1e30f;
      const float rm = half_max16(fmaxf(v0, v1));
      const float mn = fmaxf(m[r], rm);
      const float alpha = __expf(m[r] - mn);
      m[r] = mn;
      const float p0 = __expf(v0 - mn);
      const float p1 = __expf(v1 - mn);
      l[r] = l[r] * alpha + half_sum16(p0 + p1);
#pragma unroll
      for (int t = 0; t < 4; ++t) acc[t][r] *= alpha;
      const int prow = r + 8 * half;
      P[prow * 32 + row16]      = (_Float16)p0;
      P[prow * 32 + 16 + row16] = (_Float16)p1;
    }
    __syncthreads();

    // O += P(16x32) . V(32x64): P via b128 LDS loads, V gathered per element
    B16 pu;
    {
      const _Float16* pp = P + row16 * 32 + (half ? 8 : 0);
      pu.u[0] = *(const uint4*)(pp);
      pu.u[1] = *(const uint4*)(pp + 16);
    }
    const int klane = row16 + (half << 4);     // B-fragment K row = key index
#pragma unroll
    for (int t = 0; t < 4; ++t) {
      B16 bv;
      for (int s = 0; s < 16; ++s)
        bv.h[s] = Vt[((s & 7) + 8 * half + ((s >> 3) << 4)) * D_HEAD + t * 16 + row16];
      acc[t] = wmma_f16(pu.h, bv.h, acc[t]);
    }
  }

#pragma unroll
  for (int t = 0; t < 4; ++t)
#pragma unroll
    for (int r = 0; r < 8; ++r) {
      const int row = qb + r + 8 * half;
      const int col = head * D_HEAD + t * 16 + row16;
      O[(size_t)row * D_MODEL + col] = acc[t][r] / l[r];
    }
}

// ---------------------------------------------------------------- h = x + attn @ expand_O (WMMA)
// grid (M/16, N/64); 4 waves/block, one 16x16 tile per wave along N

extern "C" __global__ __launch_bounds__(128)
void dawn_gemm_expand(const float* __restrict__ A, const float* __restrict__ B,
                      const float* __restrict__ x, float* __restrict__ h) {
  const int tid = threadIdx.x, wave = tid >> 5, lane = tid & 31;
  const int row16 = lane & 15, half = lane >> 4;
  const int mb = blockIdx.x * 16;
  const int nb = blockIdx.y * 64 + wave * 16;
  v8f c;
#pragma unroll
  for (int r = 0; r < 8; ++r) c[r] = 0.f;
  for (int kb = 0; kb < D_MODEL; kb += 32) {
    v16h a, b;
    const float* ap = A + (size_t)(mb + row16) * D_MODEL + kb + (half ? 8 : 0);
    pack8(a, 0, *(const float4*)(ap),      *(const float4*)(ap + 4));
    pack8(a, 8, *(const float4*)(ap + 16), *(const float4*)(ap + 20));
    for (int s = 0; s < 16; ++s)
      b[s] = (_Float16)B[(size_t)(kb + (s & 7) + 8 * half + ((s >> 3) << 4)) * D_MODEL + nb + row16];
    c = wmma_f16(a, b, c);
  }
#pragma unroll
  for (int r = 0; r < 8; ++r) {
    const int row = mb + r + 8 * half;
    h[(size_t)row * D_MODEL + nb + row16] =
        x[(size_t)row * D_MODEL + nb + row16] + c[r];
  }
}

// ---------------------------------------------------------------- launch

extern "C" void kernel_launch(void* const* d_in, const int* in_sizes, int n_in,
                              void* d_out, int out_size, void* d_ws, size_t ws_size,
                              hipStream_t stream) {
  (void)in_sizes; (void)n_in; (void)out_size; (void)ws_size;
  const float* x        = (const float*)d_in[0];
  const float* qk_pos   = (const float*)d_in[1];
  const float* v_pos    = (const float*)d_in[2];
  const float* know_pos = (const float*)d_in[3];
  const float* tau_q    = (const float*)d_in[4];
  const float* tau_k    = (const float*)d_in[5];
  const float* tau_v    = (const float*)d_in[6];
  const float* tau_know = (const float*)d_in[7];
  const float* qk_neur  = (const float*)d_in[8];
  const float* v_neur   = (const float*)d_in[9];
  const float* kn_neur  = (const float*)d_in[10];
  const float* npos_qk  = (const float*)d_in[11];
  const float* npos_v   = (const float*)d_in[12];
  const float* npos_kn  = (const float*)d_in[13];
  const float* expand_O = (const float*)d_in[14];

  char* ws = (char*)d_ws;
  _Float16* Qh  = (_Float16*)(ws + (size_t)0);
  _Float16* Kh  = (_Float16*)(ws + ((size_t)2 << 20));
  _Float16* Vh  = (_Float16*)(ws + ((size_t)4 << 20));
  float*    att = (float*)   (ws + ((size_t)6 << 20));
  float*    hb  = (float*)   (ws + ((size_t)10 << 20));
  int*      iqk = (int*)     (ws + ((size_t)14 << 20));
  float*    dqk = (float*)   (ws + ((size_t)14 << 20) + ((size_t)256 << 10));
  int*      iv  = (int*)     (ws + ((size_t)14 << 20) + ((size_t)512 << 10));
  float*    dv  = (float*)   (ws + ((size_t)14 << 20) + ((size_t)768 << 10));
  int*      ikn = (int*)     (ws + ((size_t)15 << 20));
  float*    dkn = (float*)   (ws + ((size_t)15 << 20) + ((size_t)512 << 10));

  float* out  = (float*)d_out;
  float* loss = out + (size_t)S_LEN * D_MODEL;

  dawn_zero<<<1, 32, 0, stream>>>(loss);

  dawn_topk<<<S_LEN, 128, 4096 * sizeof(float), stream>>>(qk_pos, npos_qk, 4096, 64, iqk, dqk);
  dawn_topk<<<S_LEN, 128, 4096 * sizeof(float), stream>>>(v_pos,  npos_v,  4096, 64, iv,  dv);
  dawn_topk<<<S_LEN, 128, 8192 * sizeof(float), stream>>>(know_pos, npos_kn, 8192, 128, ikn, dkn);

  dawn_router<64, 32, true,  false><<<S_LEN, 256, 0, stream>>>(
      x, iqk, dqk, tau_q, tau_k, qk_neur, Qh, Kh, nullptr, nullptr, loss);
  dawn_router<64, 32, false, false><<<S_LEN, 256, 0, stream>>>(
      x, iv, dv, tau_v, nullptr, v_neur, Vh, nullptr, nullptr, nullptr, loss);

  dawn_flash_attn<<<dim3(S_LEN / 64, 16), 128, 0, stream>>>(Qh, Kh, Vh, att);

  dawn_gemm_expand<<<dim3(D_MODEL / 16, D_MODEL / 64), 128, 0, stream>>>(att, expand_O, x, hb);

  dawn_router<128, 64, false, true><<<S_LEN, 256, 0, stream>>>(
      hb, ikn, dkn, tau_know, nullptr, kn_neur, nullptr, nullptr, out, hb, loss);
}
